// GAT_49237505081829
// MI455X (gfx1250) — compile-verified, matrix operand order checked by reading, not run on hardware
//
#include <hip/hip_runtime.h>
#include <hip/hip_bf16.h>
#include <math.h>

#define GN 100000
#define GE 1600000
#define GIN 256
#define GH 4
#define GD 32
#define GC 47
#define GHD 128
#define GHC 188
#define NEG_SLOPE 0.2f

typedef __attribute__((ext_vector_type(16))) __bf16 v16bf;
typedef __attribute__((ext_vector_type(8)))  __bf16 v8bf;
typedef __attribute__((ext_vector_type(8)))  float  v8f;

// ---------------------------------------------------------------------------
// float atomic max via int/uint atomics (valid with -inf init)
// ---------------------------------------------------------------------------
__device__ __forceinline__ void atomicMaxF(float* addr, float val) {
    if (val >= 0.0f) {
        atomicMax((int*)addr, __float_as_int(val));
    } else {
        atomicMin((unsigned int*)addr, __float_as_uint(val));
    }
}

// ---------------------------------------------------------------------------
// fill kernel
// ---------------------------------------------------------------------------
__global__ void gat_fill(float* __restrict__ p, float v, int n) {
    int i = blockIdx.x * blockDim.x + threadIdx.x;
    if (i < n) p[i] = v;
}

// ---------------------------------------------------------------------------
// elementwise f32 -> bf16 convert (GEMM A operand)
// ---------------------------------------------------------------------------
__global__ void gat_cvt_bf16(const float* __restrict__ in,
                             __bf16* __restrict__ out, int n) {
    int i = blockIdx.x * blockDim.x + threadIdx.x;
    if (i < n) out[i] = (__bf16)in[i];
}

// ---------------------------------------------------------------------------
// W [Kin x Kout] f32 row-major -> Wt [Kout16 x Kin] bf16 (transposed,
// zero-padded columns).  Tiny matrix; cost negligible.
// ---------------------------------------------------------------------------
__global__ void gat_wt_bf16(const float* __restrict__ W,
                            __bf16* __restrict__ Wt,
                            int Kin, int Kout, int Kout16) {
    int i = blockIdx.x * blockDim.x + threadIdx.x;
    if (i >= Kout16 * Kin) return;
    int col = i / Kin, k = i % Kin;
    float v = (col < Kout) ? W[(size_t)k * Kout + col] : 0.0f;
    Wt[(size_t)col * Kin + k] = (__bf16)v;
}

// ---------------------------------------------------------------------------
// load one 16-lane-pattern bf16 fragment: two contiguous b128 loads
// per ISA 16-bit 16x32 layout (half 0: K 0-7 & 16-23; half 1: K 8-15 & 24-31)
// ---------------------------------------------------------------------------
__device__ __forceinline__ v16bf load_frag(const __bf16* base, int kb, int half) {
    v8bf lo = *(const v8bf*)(base + kb + half * 8);
    v8bf hi = *(const v8bf*)(base + kb + 16 + half * 8);
    return __builtin_shufflevector(lo, hi, 0, 1, 2, 3, 4, 5, 6, 7,
                                           8, 9, 10, 11, 12, 13, 14, 15);
}

// ---------------------------------------------------------------------------
// WMMA bf16 GEMM:  C[M x Kout](f32) = A[M x Kin](bf16) * Wt^T
// Wt is [Kout16 x Kin] bf16 (column-major weights).  One wave computes a
// 16x32 strip (2 tiles, shared A fragment).  Kout16 multiple of 32.
// ---------------------------------------------------------------------------
__global__ void gat_gemm_wmma(const __bf16* __restrict__ A,
                              const __bf16* __restrict__ Wt,
                              float* __restrict__ C,
                              int Kin, int Kout, int Kout16) {
    const int lane = threadIdx.x & 31;
    const int wave = threadIdx.x >> 5;
    const int half = lane >> 4;
    const int l16  = lane & 15;

    const int tileM  = blockIdx.x;
    const int nG     = Kout16 >> 5;            // 32-column groups
    const int group  = blockIdx.y * 4 + wave;
    if (group >= nG) return;                   // wave-uniform

    const int row  = tileM * 16 + l16;
    const int col0 = group * 32 + l16;         // tile 0 column
    const __bf16* arow = A  + (size_t)row * Kin;
    const __bf16* bc0  = Wt + (size_t)col0 * Kin;
    const __bf16* bc1  = Wt + (size_t)(col0 + 16) * Kin;

    v8f acc0 = {}, acc1 = {};

    for (int kb = 0; kb < Kin; kb += 32) {
        v16bf af  = load_frag(arow, kb, half);
        v16bf bf0 = load_frag(bc0, kb, half);
        v16bf bf1 = load_frag(bc1, kb, half);
        acc0 = __builtin_amdgcn_wmma_f32_16x16x32_bf16(
            false, af, false, bf0, (short)0, acc0, false, false);
        acc1 = __builtin_amdgcn_wmma_f32_16x16x32_bf16(
            false, af, false, bf1, (short)0, acc1, false, false);
    }

    // store: VGPR v -> M = v + 8*half, N = column
    const int c0 = col0;
    const int c1 = col0 + 16;
    #pragma unroll
    for (int v = 0; v < 8; ++v) {
        const size_t r = (size_t)(tileM * 16 + v + 8 * half) * Kout;
        if (c0 < Kout) C[r + c0] = acc0[v];
        if (c1 < Kout) C[r + c1] = acc1[v];
    }
}

// ---------------------------------------------------------------------------
// el/er: per (node, head) dot of feat row with attention vectors
// ---------------------------------------------------------------------------
__global__ void gat_elr(const float* __restrict__ feat,
                        const float* __restrict__ al,
                        const float* __restrict__ ar,
                        float* __restrict__ el, float* __restrict__ er,
                        int outD) {
    int i = blockIdx.x * blockDim.x + threadIdx.x;   // over N*H
    if (i >= GN * GH) return;
    int n = i / GH, h = i % GH;
    const float* f = feat + (size_t)n * GH * outD + h * outD;
    const float* a = al + h * outD;
    const float* b = ar + h * outD;
    float sl = 0.0f, sr = 0.0f;
    for (int d = 0; d < outD; ++d) {
        float v = f[d];
        sl += v * a[d];
        sr += v * b[d];
    }
    el[i] = sl;
    er[i] = sr;
}

// ---------------------------------------------------------------------------
// edge pass 1: segment max of leaky_relu(el[src]+er[dst]) into m[dst]
// ---------------------------------------------------------------------------
__global__ void gat_edge_max(const int* __restrict__ src,
                             const int* __restrict__ dst,
                             const float* __restrict__ el,
                             const float* __restrict__ er,
                             float* __restrict__ mb) {
    int i = blockIdx.x * blockDim.x + threadIdx.x;   // over E*H
    if (i >= GE * GH) return;
    int e = i >> 2, h = i & 3;
    float v = el[src[e] * GH + h] + er[dst[e] * GH + h];
    v = (v > 0.0f) ? v : NEG_SLOPE * v;
    atomicMaxF(&mb[dst[e] * GH + h], v);
}

// ---------------------------------------------------------------------------
// edge pass 2: ex = exp(e - m[dst]); store ex; den[dst] += ex
// ---------------------------------------------------------------------------
__global__ void gat_edge_exp(const int* __restrict__ src,
                             const int* __restrict__ dst,
                             const float* __restrict__ el,
                             const float* __restrict__ er,
                             const float* __restrict__ mb,
                             float* __restrict__ den,
                             float* __restrict__ exbuf) {
    int i = blockIdx.x * blockDim.x + threadIdx.x;   // over E*H
    if (i >= GE * GH) return;
    int e = i >> 2, h = i & 3;
    int t = dst[e] * GH + h;
    float v = el[src[e] * GH + h] + er[t];
    v = (v > 0.0f) ? v : NEG_SLOPE * v;
    float ex = __expf(v - mb[t]);
    exbuf[i] = ex;
    atomicAdd(&den[t], ex);
}

// ---------------------------------------------------------------------------
// edge pass 3: out[dst] += alpha * feat[src]; one wave per (edge, head)
// ---------------------------------------------------------------------------
__global__ void gat_edge_agg(const int* __restrict__ src,
                             const int* __restrict__ dst,
                             const float* __restrict__ exbuf,
                             const float* __restrict__ den,
                             const float* __restrict__ feat,
                             float* __restrict__ aggout,
                             int outD) {
    int gw = blockIdx.x * (blockDim.x >> 5) + (threadIdx.x >> 5);
    if (gw >= GE * GH) return;
    int lane = threadIdx.x & 31;
    int e = gw >> 2, h = gw & 3;
    int s = src[e], t = dst[e];
    float alpha = exbuf[gw] / den[t * GH + h];
    const float* fs = feat + (size_t)s * GH * outD + h * outD;
    float* ot = aggout + (size_t)t * GH * outD + h * outD;
    for (int d = lane; d < outD; d += 32) {
        atomicAdd(&ot[d], alpha * fs[d]);
    }
}

// ---------------------------------------------------------------------------
// bias + ReLU (layers 1, 2) in place over N*HD
// ---------------------------------------------------------------------------
__global__ void gat_bias_relu(float* __restrict__ h,
                              const float* __restrict__ b, int n, int F) {
    int i = blockIdx.x * blockDim.x + threadIdx.x;
    if (i >= n) return;
    float v = h[i] + b[i % F];
    h[i] = fmaxf(v, 0.0f);
}

// ---------------------------------------------------------------------------
// final: head mean + bias, then log_softmax over 47 classes.
// One wave per node: lane covers c = lane and c = lane + 32.
// ---------------------------------------------------------------------------
__global__ void gat_final(const float* __restrict__ acc3,
                          const float* __restrict__ b3,
                          float* __restrict__ out) {
    int n = blockIdx.x * (blockDim.x >> 5) + (threadIdx.x >> 5);
    if (n >= GN) return;
    int lane = threadIdx.x & 31;

    const float* base = acc3 + (size_t)n * GHC;

    int c0 = lane;            // always < 47
    int c1 = lane + 32;       // valid when lane < 15
    bool has1 = (c1 < GC);

    float v0 = 0.0f, v1 = 0.0f;
    #pragma unroll
    for (int h = 0; h < GH; ++h) {
        v0 += base[h * GC + c0] + b3[h * GC + c0];
        if (has1) v1 += base[h * GC + c1] + b3[h * GC + c1];
    }
    v0 *= 0.25f;
    v1 *= 0.25f;

    float mx = has1 ? fmaxf(v0, v1) : v0;
    #pragma unroll
    for (int o = 16; o > 0; o >>= 1) mx = fmaxf(mx, __shfl_xor(mx, o, 32));

    float se = __expf(v0 - mx) + (has1 ? __expf(v1 - mx) : 0.0f);
    #pragma unroll
    for (int o = 16; o > 0; o >>= 1) se += __shfl_xor(se, o, 32);

    float lse = mx + __logf(se);
    out[(size_t)n * GC + c0] = v0 - lse;
    if (has1) out[(size_t)n * GC + c1] = v1 - lse;
}

// ---------------------------------------------------------------------------
// launch
// ---------------------------------------------------------------------------
extern "C" void kernel_launch(void* const* d_in, const int* in_sizes, int n_in,
                              void* d_out, int out_size, void* d_ws, size_t ws_size,
                              hipStream_t stream) {
    const float* x   = (const float*)d_in[0];
    const int*   src = (const int*)d_in[1];
    const int*   dst = (const int*)d_in[2];
    const float* W1  = (const float*)d_in[3];
    const float* al1 = (const float*)d_in[4];
    const float* ar1 = (const float*)d_in[5];
    const float* b1  = (const float*)d_in[6];
    const float* W2  = (const float*)d_in[7];
    const float* al2 = (const float*)d_in[8];
    const float* ar2 = (const float*)d_in[9];
    const float* b2  = (const float*)d_in[10];
    const float* W3  = (const float*)d_in[11];
    const float* al3 = (const float*)d_in[12];
    const float* ar3 = (const float*)d_in[13];
    const float* b3  = (const float*)d_in[14];
    float* out = (float*)d_out;

    // workspace carve-up
    float* ws    = (float*)d_ws;
    float* feat  = ws;                              // N*188
    float* acc3  = feat  + (size_t)GN * GHC;        // N*188
    float* hbuf  = acc3  + (size_t)GN * GHC;        // N*128
    float* exbuf = hbuf  + (size_t)GN * GHD;        // E*H
    float* el    = exbuf + (size_t)GE * GH;         // N*H
    float* er    = el    + (size_t)GN * GH;
    float* mb    = er    + (size_t)GN * GH;
    float* den   = mb    + (size_t)GN * GH;
    __bf16* abf  = (__bf16*)(den + (size_t)GN * GH);        // N*256 bf16
    __bf16* wtbf = abf + (size_t)GN * GIN;                  // 192*256 bf16

    const int NH  = GN * GH;
    const int EH  = GE * GH;
    const int T   = 256;
    const dim3 blkEH((EH + T - 1) / T);
    const dim3 blkNH((NH + T - 1) / T);
    const dim3 blkAgg((EH + 7) / 8);                // 8 waves / block
    const float NEGINF = -__builtin_huge_valf();

    struct Layer {
        const float* hin; const float* W; const float* al; const float* ar;
        const float* b;  float* aggout; int Kin; int Kout; int outD;
    } L[3] = {
        { x,    W1, al1, ar1, b1, hbuf, GIN, GHD, GD },
        { hbuf, W2, al2, ar2, b2, hbuf, GHD, GHD, GD },
        { hbuf, W3, al3, ar3, b3, acc3, GHD, GHC, GC },
    };

    for (int l = 0; l < 3; ++l) {
        const int Kin    = L[l].Kin;
        const int Kout   = L[l].Kout;
        const int outD   = L[l].outD;
        const int Kout16 = ((Kout + 31) / 32) * 32;   // pad to 32 columns
        const int nG     = Kout16 / 32;

        // 0) convert GEMM operands to bf16 (A elementwise, W transposed+padded)
        {
            int na = GN * Kin;
            hipLaunchKernelGGL(gat_cvt_bf16, dim3((na + T - 1) / T), dim3(T), 0,
                               stream, L[l].hin, abf, na);
            int nw = Kout16 * Kin;
            hipLaunchKernelGGL(gat_wt_bf16, dim3((nw + T - 1) / T), dim3(T), 0,
                               stream, L[l].W, wtbf, Kin, Kout, Kout16);
        }

        // 1) dense GEMM via WMMA bf16: feat = hin @ W
        dim3 gg(GN / 16, (nG + 3) / 4);
        hipLaunchKernelGGL(gat_gemm_wmma, gg, dim3(128), 0, stream,
                           abf, wtbf, feat, Kin, Kout, Kout16);

        // 2) attention logits
        hipLaunchKernelGGL(gat_elr, blkNH, dim3(T), 0, stream,
                           feat, L[l].al, L[l].ar, el, er, outD);

        // 3) init segment buffers + aggregation output
        hipLaunchKernelGGL(gat_fill, blkNH, dim3(T), 0, stream, mb, NEGINF, NH);
        hipLaunchKernelGGL(gat_fill, blkNH, dim3(T), 0, stream, den, 0.0f, NH);
        {
            int n = GN * Kout;
            hipLaunchKernelGGL(gat_fill, dim3((n + T - 1) / T), dim3(T), 0, stream,
                               L[l].aggout, 0.0f, n);
        }

        // 4) segment max
        hipLaunchKernelGGL(gat_edge_max, blkEH, dim3(T), 0, stream,
                           src, dst, el, er, mb);
        // 5) exp + segment sum
        hipLaunchKernelGGL(gat_edge_exp, blkEH, dim3(T), 0, stream,
                           src, dst, el, er, mb, den, exbuf);
        // 6) weighted scatter-aggregate
        hipLaunchKernelGGL(gat_edge_agg, blkAgg, dim3(T), 0, stream,
                           src, dst, exbuf, den, feat, L[l].aggout, outD);

        // 7) epilogue
        if (l < 2) {
            int n = GN * GHD;
            hipLaunchKernelGGL(gat_bias_relu, dim3((n + T - 1) / T), dim3(T), 0,
                               stream, hbuf, L[l].b, n, GHD);
        }
    }

    // head mean + log_softmax -> d_out
    hipLaunchKernelGGL(gat_final, dim3((GN + 7) / 8), dim3(T), 0, stream,
                       acc3, b3, out);
    (void)in_sizes; (void)n_in; (void)out_size; (void)ws_size;
}
